// Cell_26568667693298
// MI455X (gfx1250) — matrix operand order repaired and checked by reading, not compiled, not run on hardware
//
#include <hip/hip_runtime.h>

// ---------------- problem constants (fixed by setup_inputs) ----------------
#define Bb    128
#define Hh    40
#define Wimg  40
#define Cch   16
#define Ss    32
#define NPIX  (Hh * Wimg)      // 1600
#define HID   128
#define PKp   64               // perception K padded 48 -> 64
#define NT    (NPIX / 16)      // 100 M-tiles of 16 pixels
#define NW    8                // waves per block (wave32)
#define TPB   256

typedef __attribute__((ext_vector_type(16))) _Float16 v16h;
typedef __attribute__((ext_vector_type(8)))  _Float16 v8h;
typedef __attribute__((ext_vector_type(8)))  float    v8f;

static __device__ __forceinline__ v16h hcat(v8h lo, v8h hi) {
  return __builtin_shufflevector(lo, hi, 0,1,2,3,4,5,6,7,8,9,10,11,12,13,14,15);
}

static __device__ __forceinline__ v8f wmma_f16(v16h a, v16h b, v8f c) {
  // D(16x16,f32) = A(16x32,f16) x B(32x16,f16) + C
  return __builtin_amdgcn_wmma_f32_16x16x32_f16(false, a, false, b, (short)0, c,
                                                false, false);
}

// A fragment (16x32 f16, MxK). Row-major tile in LDS, row stride `stride`
// halves. lane = M (mod 16); lanes<16 hold K [kb..kb+7],[kb+16..kb+23];
// lanes>=16 hold K [kb+8..],[kb+24..]  (ISA 7.12.2 16-bit A layout).
static __device__ __forceinline__ v16h loadA(const _Float16* tile, int stride,
                                             int lane, int kbase) {
  const int m  = lane & 15;
  const int ko = kbase + ((lane & 16) ? 8 : 0);
  const _Float16* p = tile + m * stride + ko;
  v8h lo = *(const v8h*)(p);        // ds_load_b128
  v8h hi = *(const v8h*)(p + 16);   // ds_load_b128
  return hcat(lo, hi);
}

// B fragment (32x16 f16, KxN). wrow points at this lane's N-column stored as a
// row of K values (weights kept as [out][K] in LDS). lanes<16: K [kb..kb+7],
// [kb+16..kb+23]; lanes>=16: K [kb+8..kb+15],[kb+24..kb+31].
static __device__ __forceinline__ v16h loadB(const _Float16* wrow, int lane,
                                             int kbase) {
  const _Float16* p = wrow + kbase + ((lane & 16) ? 8 : 0);
  v8h lo = *(const v8h*)(p);
  v8h hi = *(const v8h*)(p + 16);
  return hcat(lo, hi);
}

static __device__ __forceinline__ float clamp01(float v) {
  return fminf(fmaxf(v, 0.f), 1.f);
}

__global__ __launch_bounds__(TPB, 1)
void nca_kernel(const float* __restrict__ init,
                const float* __restrict__ W1,
                const float* __restrict__ b1,
                const float* __restrict__ W2,
                const int*   __restrict__ updMask,
                float* __restrict__ rgbOut,
                float* __restrict__ stacked) {
  const int b    = blockIdx.x;          // one block per batch image
  const int tid  = threadIdx.x;
  const int wave = tid >> 5;
  const int lane = tid & 31;

  __shared__ _Float16 maskedL[NPIX * Cch];   // 50 KB  masked state (f16)
  __shared__ float    alphaL[NPIX];          // 6.25KB alpha plane
  __shared__ _Float16 W1L[HID * PKp];        // 16 KB  W1 as [o][K], K pad 64
  __shared__ _Float16 W2L[Cch * HID];        // 4 KB   W2 as [j][K]
  __shared__ float    b1L[HID];              // 0.5KB
  __shared__ _Float16 pT[NW][16 * PKp];      // 16 KB  per-wave perception tile
  __shared__ _Float16 hT[NW][16 * HID];      // 32 KB  per-wave hidden tile

  // ---- stage weights to LDS (f16), pad K 48->64 with zeros ----
  for (int i = tid; i < HID * PKp; i += TPB) {
    const int o = i >> 6, k = i & 63;
    W1L[i] = (k < 48) ? (_Float16)W1[o * 48 + k] : (_Float16)0.f;
  }
  for (int i = tid; i < Cch * HID; i += TPB) W2L[i] = (_Float16)W2[i];
  for (int i = tid; i < HID; i += TPB)       b1L[i] = b1[i];

  // ---- frame 0: copy initial state into stacked[0], emit rgb[0], seed alpha
  const float* st0  = init    + (size_t)b * NPIX * Cch;
  float*       stk0 = stacked + (size_t)b * NPIX * Cch;
  for (int i = tid; i < NPIX * Cch; i += TPB) stk0[i] = st0[i];
  {
    float* rgb0 = rgbOut + (size_t)b * NPIX * 3;
    for (int p = tid; p < NPIX; p += TPB) {
      const float av = st0[p * Cch + 3];
      alphaL[p] = av;
      const float a = clamp01(av);
#pragma unroll
      for (int ch = 0; ch < 3; ++ch)
        rgb0[p * 3 + ch] =
            truncf(clamp01(1.f - a + st0[p * Cch + ch]) * 255.f);
    }
  }
  __syncthreads();

  const size_t frameStride = (size_t)Bb * NPIX * Cch;
  const size_t rgbStride   = (size_t)Bb * NPIX * 3;

  auto tapf = [&](int yy, int xx, int c) -> float {
    return (yy >= 0 && yy < Hh && xx >= 0 && xx < Wimg)
               ? (float)maskedL[(yy * Wimg + xx) * Cch + c]
               : 0.f;
  };

  for (int s = 0; s < Ss; ++s) {
    const float* cur = stacked + (size_t)s * frameStride + (size_t)b * NPIX * Cch;
    float*       nxt = stacked + (size_t)(s + 1) * frameStride + (size_t)b * NPIX * Cch;
    const int*   msk = updMask + ((size_t)s * Bb + b) * NPIX;

    // ---- Phase A: masked = state * alive(alpha 3x3 maxpool >= 0.1) ----
    for (int p = tid; p < NPIX; p += TPB) {
      const int y = p / Wimg, x = p % Wimg;
      float mx = -1e30f;
#pragma unroll
      for (int dy = -1; dy <= 1; ++dy)
#pragma unroll
        for (int dx = -1; dx <= 1; ++dx) {
          const int yy = y + dy, xx = x + dx;
          if (yy >= 0 && yy < Hh && xx >= 0 && xx < Wimg)
            mx = fmaxf(mx, alphaL[yy * Wimg + xx]);
        }
      const float am = (mx >= 0.1f) ? 1.f : 0.f;
#pragma unroll
      for (int c = 0; c < Cch; ++c)
        maskedL[p * Cch + c] = (_Float16)(cur[p * Cch + c] * am);
    }
    __syncthreads();

    // ---- Phase B: per-wave 16-pixel tiles, MLP via WMMA ----
    const int nN   = lane & 15;
    const int mOff = (lane & 16) ? 8 : 0;
    for (int t = wave; t < NT; t += NW) {
      const int p0 = t * 16;

      // build perception tile [16 pixels][64 K]: K0..15 masked, 16..31 sx,
      // 32..47 sy, 48..63 zero. lane%16 = pixel, lane/16 = channel half.
      {
        const int p  = p0 + (lane & 15);
        const int y  = p / Wimg, x = p % Wimg;
        const int cb = (lane & 16) ? 8 : 0;
        _Float16* row = &pT[wave][(lane & 15) * PKp];
        v8h z = {};
        *(v8h*)(row + 48 + ((lane & 16) ? 8 : 0)) = z;  // zero K-pad
#pragma unroll
        for (int cc = 0; cc < 8; ++cc) {
          const int c = cb + cc;
          const float t00 = tapf(y - 1, x - 1, c);
          const float t01 = tapf(y - 1, x,     c);
          const float t02 = tapf(y - 1, x + 1, c);
          const float t10 = tapf(y,     x - 1, c);
          const float t12 = tapf(y,     x + 1, c);
          const float t20 = tapf(y + 1, x - 1, c);
          const float t21 = tapf(y + 1, x,     c);
          const float t22 = tapf(y + 1, x + 1, c);
          const float ctr = (float)maskedL[p * Cch + c];
          const float sx = (-t00 + t02 - 2.f * t10 + 2.f * t12 - t20 + t22) * 0.125f;
          const float sy = (-t00 - 2.f * t01 - t02 + t20 + 2.f * t21 + t22) * 0.125f;
          row[c]      = (_Float16)ctr;
          row[16 + c] = (_Float16)sx;
          row[32 + c] = (_Float16)sy;
        }
      }

      // GEMM1: H(16x128) = P(16x64) * W1^T, K = 2 chunks of 32
      const v16h a0 = loadA(pT[wave], PKp, lane, 0);
      const v16h a1 = loadA(pT[wave], PKp, lane, 32);
#pragma unroll
      for (int n = 0; n < 8; ++n) {
        const int o = n * 16 + nN;
        v8f acc = {};
        acc = wmma_f16(a0, loadB(&W1L[o * PKp], lane, 0),  acc);
        acc = wmma_f16(a1, loadB(&W1L[o * PKp], lane, 32), acc);
        const float bias = b1L[o];
#pragma unroll
        for (int r = 0; r < 8; ++r) {    // D: VGPR r -> M = r + mOff
          const float v = acc[r] + bias;
          hT[wave][(r + mOff) * HID + o] = (_Float16)(v > 0.f ? v : 0.f);
        }
      }

      // GEMM2: U(16x16) = relu(H)(16x128) * W2^T, K = 4 chunks of 32
      v8f acc2 = {};
#pragma unroll
      for (int kc = 0; kc < 4; ++kc) {
        const v16h af = loadA(hT[wave], HID, lane, kc * 32);
        const v16h bf = loadB(&W2L[nN * HID], lane, kc * 32);
        acc2 = wmma_f16(af, bf, acc2);
      }

      // tmp_new = state + update * upd_mask ; stash tmp alpha for Phase C
#pragma unroll
      for (int r = 0; r < 8; ++r) {
        const int   p    = p0 + r + mOff;
        const float mval = (float)msk[p];
        const float v    = cur[p * Cch + nN] + acc2[r] * mval;
        nxt[p * Cch + nN] = v;
        if (nN == 3) alphaL[p] = v;
      }
    }
    __syncthreads();

    // ---- Phase C: final alive mask on tmp new state + rgb frame s+1 ----
    float keep[7];
    int   cnt = 0;
    float* rgbF = rgbOut + (size_t)(s + 1) * rgbStride + (size_t)b * NPIX * 3;
    for (int p = tid; p < NPIX; p += TPB) {
      const int y = p / Wimg, x = p % Wimg;
      float mx = -1e30f;
#pragma unroll
      for (int dy = -1; dy <= 1; ++dy)
#pragma unroll
        for (int dx = -1; dx <= 1; ++dx) {
          const int yy = y + dy, xx = x + dx;
          if (yy >= 0 && yy < Hh && xx >= 0 && xx < Wimg)
            mx = fmaxf(mx, alphaL[yy * Wimg + xx]);
        }
      const float am2 = (mx >= 0.1f) ? 1.f : 0.f;
      const float na  = alphaL[p] * am2;   // final alpha
      keep[cnt++] = na;
#pragma unroll
      for (int c = 0; c < Cch; ++c) nxt[p * Cch + c] *= am2;
      const float a = clamp01(na);
#pragma unroll
      for (int ch = 0; ch < 3; ++ch)
        rgbF[p * 3 + ch] =
            truncf(clamp01(1.f - a + nxt[p * Cch + ch]) * 255.f);
    }
    __syncthreads();
    cnt = 0;
    for (int p = tid; p < NPIX; p += TPB) alphaL[p] = keep[cnt++];
    __syncthreads();
  }
}

extern "C" void kernel_launch(void* const* d_in, const int* in_sizes, int n_in,
                              void* d_out, int out_size, void* d_ws,
                              size_t ws_size, hipStream_t stream) {
  (void)in_sizes; (void)n_in; (void)out_size; (void)d_ws; (void)ws_size;
  const float* init = (const float*)d_in[0];
  const float* W1   = (const float*)d_in[1];
  const float* b1   = (const float*)d_in[2];
  const float* W2   = (const float*)d_in[3];
  const int*   um   = (const int*)d_in[4];

  float* rgb     = (float*)d_out;                       // (33,128,40,40,3)
  float* stacked = rgb + (size_t)(Ss + 1) * Bb * NPIX * 3;  // (33,128,40,40,16)

  hipLaunchKernelGGL(nca_kernel, dim3(Bb), dim3(TPB), 0, stream,
                     init, W1, b1, W2, um, rgb, stacked);
}
